// Interaction_80685255623032
// MI455X (gfx1250) — matrix operand order repaired and checked by reading, not compiled, not run on hardware
//
#include <hip/hip_runtime.h>
#include <hip/hip_bf16.h>

// ---------------------------------------------------------------------------
// Types for CDNA5 WMMA (wave32)
// ---------------------------------------------------------------------------
typedef __bf16 bf16_t;
typedef __bf16 v16bf __attribute__((ext_vector_type(16)));
typedef float  v8f   __attribute__((ext_vector_type(8)));

#define BM 128
#define BN 128
#define BK 32

// ---------------------------------------------------------------------------
// prep: per-row dot(E[row,:], Wd) -> dots[row]; also bf16 conversions:
//   Ebf = bf16(E), and optionally Ewbf = bf16(E * Wm)
// One wave (32 lanes) per row, 8 rows per 256-thread block.
// ---------------------------------------------------------------------------
template <bool HASW>
__global__ __launch_bounds__(256) void prep_kernel(
    const float* __restrict__ E, const float* __restrict__ Wd,
    const float* __restrict__ Wm, float* __restrict__ dots,
    bf16_t* __restrict__ Ebf, bf16_t* __restrict__ Ewbf, int H)
{
    int row  = blockIdx.x * 8 + (threadIdx.x >> 5);
    int lane = threadIdx.x & 31;
    const float* src = E + (size_t)row * H;
    float partial = 0.0f;
    for (int h = lane; h < H; h += 32) {
        float v = src[h];
        partial += v * Wd[h];
        Ebf[(size_t)row * H + h] = (bf16_t)v;
        if (HASW) Ewbf[(size_t)row * H + h] = (bf16_t)(v * Wm[h]);
    }
    #pragma unroll
    for (int off = 16; off > 0; off >>= 1)
        partial += __shfl_xor(partial, off, 32);
    if (lane == 0) dots[row] = partial;
}

// ---------------------------------------------------------------------------
// Tiled bf16 GEMM:  C[M,N](f32) = A[M,K](bf16,row-major) x B
//   BT = true : B stored as (N,K) row-major (i.e. we multiply by B^T layout)
//   BT = false: B stored as (K,N) row-major
//   EPI       : C += rowBias[m] + colBias[n]; mask<0.5 -> -1e9
//   WRBF      : also write bf16 copy of C
// Grid: (N/BN, M/BM, batch). Block: 256 threads = 8 waves (wave32).
// Wave (wm in 0..1, wn in 0..3) computes a 64x32 patch = 4x2 WMMA 16x16 tiles.
// LDS B tile is kept transposed (sBt[n][k]) so each lane's B fragment is 16
// contiguous bf16 -> 2x ds_load_b128 (same as the A fragment path).
// Requires M%128==0, N%128==0, K%32==0 (true for all calls here).
// ---------------------------------------------------------------------------
template <bool BT, bool EPI, bool WRBF>
__global__ __launch_bounds__(256) void gemm_bf16_kernel(
    const bf16_t* __restrict__ A, const bf16_t* __restrict__ Bm,
    float* __restrict__ C, bf16_t* __restrict__ Cbf,
    int M, int N, int K,
    const float* __restrict__ rowBias, const float* __restrict__ colBias,
    const float* __restrict__ rowMask, const float* __restrict__ colMask)
{
    const int b = blockIdx.z;
    A  += (size_t)b * M * K;
    Bm += (size_t)b * N * K;
    C  += (size_t)b * M * N;
    if (WRBF) Cbf += (size_t)b * M * N;
    if (EPI) {
        rowBias += (size_t)b * M;
        colBias += (size_t)b * N;
        rowMask += (size_t)b * M;
        colMask += (size_t)b * N;
    }

    __shared__ bf16_t sA [BM][BK + 8];   // [m][k]
    __shared__ bf16_t sBt[BN][BK + 8];   // [n][k]  (transposed B tile)

    const int tid  = threadIdx.x;
    const int lane = tid & 31;
    const int wave = tid >> 5;
    const int wm   = wave >> 2;   // 0..1 -> 64-row group
    const int wn   = wave & 3;    // 0..3 -> 32-col group
    const int m0   = blockIdx.y * BM;
    const int n0   = blockIdx.x * BN;

    v8f acc[4][2];
    #pragma unroll
    for (int i = 0; i < 4; ++i)
        #pragma unroll
        for (int j = 0; j < 2; ++j) acc[i][j] = (v8f)0.0f;

    const int lrow  = lane & 15;
    const int ahalf = (lane >> 4) * 8;    // unused; A frag reads 16 contiguous K at bhalf
    const int bhalf = (lane >> 4) * 16;   // B-frag K-half base (contiguous 16)

    for (int k0 = 0; k0 < K; k0 += BK) {
        // ---- stage A tile (128x32): thread -> 16 contiguous bf16 ----
        {
            int r = tid >> 1;
            int c = (tid & 1) * 16;
            const bf16_t* src = A + (size_t)(m0 + r) * K + k0 + c;
            if (k0 + BK < K) __builtin_prefetch(src + BK, 0, 1); // global_prefetch_b8
            #pragma unroll
            for (int e = 0; e < 16; ++e) sA[r][c + e] = src[e];
        }
        // ---- stage B tile transposed into sBt[n][k] ----
        if (BT) {
            // B is (N,K): straight contiguous copy per n-row
            int n = tid >> 1;
            int c = (tid & 1) * 16;
            const bf16_t* src = Bm + (size_t)(n0 + n) * K + k0 + c;
            #pragma unroll
            for (int e = 0; e < 16; ++e) sBt[n][c + e] = src[e];
        } else {
            // B is (K,N): transpose at staging time; pack two K-adjacent
            // bf16 into one 32-bit LDS store (k even, row pitch 80B -> 4B ok)
            int kk2 = (tid >> 4) * 2;       // 0,2,..,30
            int ng  = (tid & 15) * 8;       // 0,8,..,120
            const unsigned short* s0 =
                (const unsigned short*)(Bm + (size_t)(k0 + kk2) * N + n0 + ng);
            const unsigned short* s1 =
                (const unsigned short*)(Bm + (size_t)(k0 + kk2 + 1) * N + n0 + ng);
            #pragma unroll
            for (int e = 0; e < 8; ++e) {
                unsigned int pk = (unsigned int)s0[e] | ((unsigned int)s1[e] << 16);
                *(unsigned int*)&sBt[ng + e][kk2] = pk;
            }
        }
        __syncthreads();

        // ---- B fragments: lane = col, 16 contiguous K -> 2x ds_load_b128 ----
        v16bf bfrag[2];
        #pragma unroll
        for (int j = 0; j < 2; ++j) {
            const bf16_t* bp = &sBt[wn * 32 + j * 16 + lrow][bhalf];
            #pragma unroll
            for (int e = 0; e < 16; ++e) bfrag[j][e] = bp[e];
        }
        // ---- A fragments (two 8-element runs -> 2x ds_load_b128) + WMMA ----
        #pragma unroll
        for (int i = 0; i < 4; ++i) {
            const bf16_t* ap = &sA[wm * 64 + i * 16 + lrow][0];
            v16bf afrag;
            #pragma unroll
            for (int e = 0; e < 16; ++e)
                afrag[e] = ap[bhalf + e];
            #pragma unroll
            for (int j = 0; j < 2; ++j)
                acc[i][j] = __builtin_amdgcn_wmma_f32_16x16x32_bf16(
                    false, afrag, false, bfrag[j], (short)0, acc[i][j],
                    false, false);
        }
        __syncthreads();
    }

    // ---- epilogue: C/D layout: row = (lane>>4)*8 + r, col = lane&15 ----
    const int lcol   = lane & 15;
    const int rshift = (lane >> 4) * 8;
    #pragma unroll
    for (int i = 0; i < 4; ++i) {
        #pragma unroll
        for (int j = 0; j < 2; ++j) {
            int nt = n0 + wn * 32 + j * 16 + lcol;
            float cb = 0.0f, cm = 1.0f;
            if (EPI) { cb = colBias[nt]; cm = colMask[nt]; }
            #pragma unroll
            for (int r = 0; r < 8; ++r) {
                int mm  = m0 + wm * 64 + i * 16 + rshift + r;
                float v = acc[i][j][r];
                if (EPI) {
                    v += rowBias[mm] + cb;
                    if (rowMask[mm] * cm < 0.5f) v = -1000000000.0f;
                }
                C[(size_t)mm * N + nt] = v;
                if (WRBF) Cbf[(size_t)mm * N + nt] = (bf16_t)v;
            }
        }
    }
}

// ---------------------------------------------------------------------------
// Row softmax stats: one block per (b,p) row of U (contiguous N floats)
// ---------------------------------------------------------------------------
__global__ __launch_bounds__(256) void row_stats_kernel(
    const float* __restrict__ U, float* __restrict__ rmax,
    float* __restrict__ rsum, int N)
{
    const float* row = U + (size_t)blockIdx.x * N;
    __shared__ float red[256];
    float m = -INFINITY;
    for (int i = threadIdx.x; i < N; i += 256) m = fmaxf(m, row[i]);
    red[threadIdx.x] = m;
    __syncthreads();
    for (int s = 128; s > 0; s >>= 1) {
        if (threadIdx.x < s)
            red[threadIdx.x] = fmaxf(red[threadIdx.x], red[threadIdx.x + s]);
        __syncthreads();
    }
    m = red[0];
    __syncthreads();
    float s = 0.0f;
    for (int i = threadIdx.x; i < N; i += 256) s += __expf(row[i] - m);
    red[threadIdx.x] = s;
    __syncthreads();
    for (int t = 128; t > 0; t >>= 1) {
        if (threadIdx.x < t) red[threadIdx.x] += red[threadIdx.x + t];
        __syncthreads();
    }
    if (threadIdx.x == 0) { rmax[blockIdx.x] = m; rsum[blockIdx.x] = red[0]; }
}

// ---------------------------------------------------------------------------
// Column softmax stats: block = (q-tile of 64, batch); coalesced over q.
// ---------------------------------------------------------------------------
__global__ __launch_bounds__(256) void col_stats_kernel(
    const float* __restrict__ U, float* __restrict__ cmax,
    float* __restrict__ csum, int P, int Q)
{
    int b  = blockIdx.y;
    int q0 = blockIdx.x * 64;
    int qi = threadIdx.x & 63;
    int pg = threadIdx.x >> 6; // 0..3
    const float* Ub = U + (size_t)b * P * Q;
    __shared__ float red[4][64];
    float m = -INFINITY;
    for (int p = pg; p < P; p += 4)
        m = fmaxf(m, Ub[(size_t)p * Q + q0 + qi]);
    red[pg][qi] = m;
    __syncthreads();
    if (pg == 0) {
        m = fmaxf(fmaxf(red[0][qi], red[1][qi]), fmaxf(red[2][qi], red[3][qi]));
        red[0][qi] = m;
    }
    __syncthreads();
    m = red[0][qi];
    float s = 0.0f;
    for (int p = pg; p < P; p += 4)
        s += __expf(Ub[(size_t)p * Q + q0 + qi] - m);
    __syncthreads();
    red[pg][qi] = s;
    __syncthreads();
    if (pg == 0) {
        s = red[0][qi] + red[1][qi] + red[2][qi] + red[3][qi];
        cmax[(size_t)b * Q + q0 + qi] = m;
        csum[(size_t)b * Q + q0 + qi] = s;
    }
}

// ---------------------------------------------------------------------------
// Normalize 64x64 tile: emit A_p (bf16, [p][q]) and B_p^T (bf16, [q][p],
// via LDS transpose so the next GEMMs read a row-major A operand).
// ---------------------------------------------------------------------------
__global__ __launch_bounds__(256) void normalize_kernel(
    const float* __restrict__ U,
    const float* __restrict__ rmax, const float* __restrict__ rsum,
    const float* __restrict__ cmax, const float* __restrict__ csum,
    bf16_t* __restrict__ Ap, bf16_t* __restrict__ BpT, int P, int Q)
{
    int b  = blockIdx.z;
    int p0 = blockIdx.y * 64;
    int q0 = blockIdx.x * 64;
    const float* Ub = U + (size_t)b * P * Q;
    __shared__ float sT[64][65];
    int r  = threadIdx.x >> 2;        // 0..63
    int cg = (threadIdx.x & 3) * 16;  // 0,16,32,48
    int p  = p0 + r;
    float rm = rmax[(size_t)b * P + p];
    float rs = rsum[(size_t)b * P + p];
    #pragma unroll
    for (int e = 0; e < 16; ++e) {
        int q = q0 + cg + e;
        float u = Ub[(size_t)p * Q + q];
        Ap[(size_t)b * P * Q + (size_t)p * Q + q] = (bf16_t)(__expf(u - rm) / rs);
        sT[cg + e][r] = __expf(u - cmax[(size_t)b * Q + q]) / csum[(size_t)b * Q + q];
    }
    __syncthreads();
    int q = q0 + r;
    #pragma unroll
    for (int e = 0; e < 16; ++e)
        BpT[(size_t)b * (size_t)Q * P + (size_t)q * P + p0 + cg + e] =
            (bf16_t)sT[r][cg + e];
}

// ---------------------------------------------------------------------------
// Assemble G = concat([E, X1, X2, E*X1, E*X2], axis=-1)
// ---------------------------------------------------------------------------
__global__ __launch_bounds__(256) void assemble_kernel(
    const float* __restrict__ E, const float* __restrict__ X1,
    const float* __restrict__ X2, float* __restrict__ G, int H)
{
    size_t idx = (size_t)blockIdx.x * 256 + threadIdx.x;
    size_t row = idx / (size_t)H;
    int    h   = (int)(idx % (size_t)H);
    float e = E[idx], x1 = X1[idx], x2 = X2[idx];
    float* g = G + row * (size_t)(5 * H) + h;
    g[0]     = e;
    g[H]     = x1;
    g[2 * H] = x2;
    g[3 * H] = e * x1;
    g[4 * H] = e * x2;
}

// ---------------------------------------------------------------------------
// Host launcher
// ---------------------------------------------------------------------------
extern "C" void kernel_launch(void* const* d_in, const int* in_sizes, int n_in,
                              void* d_out, int out_size, void* d_ws, size_t ws_size,
                              hipStream_t stream)
{
    (void)in_sizes; (void)n_in; (void)out_size; (void)ws_size;
    constexpr int B = 4, SQ = 2048, SP = 2048, H = 256;

    const float* Eq    = (const float*)d_in[0]; // (B,SQ,H)
    const float* Ep    = (const float*)d_in[1]; // (B,SP,H)
    const float* mask1 = (const float*)d_in[2]; // (B,SP) -> p dim
    const float* mask2 = (const float*)d_in[3]; // (B,SQ) -> q dim
    const float* W     = (const float*)d_in[4]; // (3H,)
    float* out = (float*)d_out;

    char* ws = (char*)d_ws;
    size_t off = 0;
    auto alloc = [&](size_t bytes) -> void* {
        off = (off + 255) & ~(size_t)255;
        void* p = ws + off;
        off += bytes;
        return p;
    };

    const size_t NE = (size_t)B * SQ * H;   // == B*SP*H
    const size_t NU = (size_t)B * SP * SQ;

    bf16_t* Eqbf  = (bf16_t*)alloc(NE * 2);
    bf16_t* Epbf  = (bf16_t*)alloc(NE * 2);
    bf16_t* Epwbf = (bf16_t*)alloc(NE * 2);
    float*  sqv   = (float*)alloc((size_t)B * SQ * 4);
    float*  spv   = (float*)alloc((size_t)B * SP * 4);
    float*  U     = (float*)alloc(NU * 4);
    float*  rmax  = (float*)alloc((size_t)B * SP * 4);
    float*  rsum  = (float*)alloc((size_t)B * SP * 4);
    float*  cmax  = (float*)alloc((size_t)B * SQ * 4);
    float*  csum  = (float*)alloc((size_t)B * SQ * 4);
    bf16_t* Apbf  = (bf16_t*)alloc(NU * 2);
    bf16_t* BpTbf = (bf16_t*)alloc(NU * 2);
    float*  A1f   = (float*)alloc(NE * 4);
    float*  B1f   = (float*)alloc(NE * 4);
    float*  A2f   = (float*)alloc(NE * 4);
    float*  B2f   = (float*)alloc(NE * 4);
    bf16_t* A1bf  = (bf16_t*)alloc(NE * 2);
    bf16_t* B1bf  = (bf16_t*)alloc(NE * 2);

    // 1) prep: sq + Eq->bf16 ; sp + Ep->bf16 + (Ep*w_m)->bf16
    prep_kernel<false><<<B * SQ / 8, 256, 0, stream>>>(
        Eq, W, nullptr, sqv, Eqbf, nullptr, H);
    prep_kernel<true><<<B * SP / 8, 256, 0, stream>>>(
        Ep, W + H, W + 2 * H, spv, Epbf, Epwbf, H);

    // 2) U = (Ep*w_m) @ Eq^T + sp + sq, masked   (M=SP, N=SQ, K=H; B is (N,K))
    gemm_bf16_kernel<true, true, false><<<dim3(SQ / BN, SP / BM, B), 256, 0, stream>>>(
        Epwbf, Eqbf, U, nullptr, SP, SQ, H, spv, sqv, mask1, mask2);

    // 3) softmax statistics (U resident in 192MB L2)
    row_stats_kernel<<<B * SP, 256, 0, stream>>>(U, rmax, rsum, SQ);
    col_stats_kernel<<<dim3(SQ / 64, B), 256, 0, stream>>>(U, cmax, csum, SP, SQ);

    // 4) A_p (bf16) and B_p^T (bf16, transposed)
    normalize_kernel<<<dim3(SQ / 64, SP / 64, B), 256, 0, stream>>>(
        U, rmax, rsum, cmax, csum, Apbf, BpTbf, SP, SQ);

    // 5) A1 = A_p @ E_q  (M=SP,N=H,K=SQ; B is (K,N))   -> fp32 + bf16 copy
    gemm_bf16_kernel<false, false, true><<<dim3(H / BN, SP / BM, B), 256, 0, stream>>>(
        Apbf, Eqbf, A1f, A1bf, SP, H, SQ, nullptr, nullptr, nullptr, nullptr);
    //    B1 = B_p^T @ E_p (M=SQ,N=H,K=SP)
    gemm_bf16_kernel<false, false, true><<<dim3(H / BN, SQ / BM, B), 256, 0, stream>>>(
        BpTbf, Epbf, B1f, B1bf, SQ, H, SP, nullptr, nullptr, nullptr, nullptr);
    // 6) A2 = A_p @ B1 ; B2 = B_p^T @ A1
    gemm_bf16_kernel<false, false, false><<<dim3(H / BN, SP / BM, B), 256, 0, stream>>>(
        Apbf, B1bf, A2f, nullptr, SP, H, SQ, nullptr, nullptr, nullptr, nullptr);
    gemm_bf16_kernel<false, false, false><<<dim3(H / BN, SQ / BM, B), 256, 0, stream>>>(
        BpTbf, A1bf, B2f, nullptr, SQ, H, SP, nullptr, nullptr, nullptr, nullptr);

    // 7) outputs: G_p_q = [E_q,B1,B2,E_q*B1,E_q*B2] then G_q_p = [E_p,A1,A2,...]
    assemble_kernel<<<(unsigned)(NE / 256), 256, 0, stream>>>(Eq, B1f, B2f, out, H);
    assemble_kernel<<<(unsigned)(NE / 256), 256, 0, stream>>>(
        Ep, A1f, A2f, out + (size_t)B * SQ * 5 * H, H);
}
